// DMPNN_35923106464040
// MI455X (gfx1250) — compile-verified, matrix operand order checked by reading, not run on hardware
//
#include <hip/hip_runtime.h>
#include <hip/hip_bf16.h>
#include <stdint.h>

// ---------------------------------------------------------------------------
// DMPNN for MI455X (gfx1250, wave32).
//   E=8192 edges, N=4096 nodes, NODE_D=64, EDGE_D=16, H=256, T=4.
// Strategy:
//   * adjacency matmuls -> one-time CSR extraction + ordered gather-sum + fused LN
//   * dense weight GEMMs -> v_wmma_f32_16x16x32_bf16; weights pre-transposed to
//     [N][K] bf16 and A panels pre-converted to bf16 so every tile operand is
//     two global_load_b128 per lane feeding the WMMA pipe (no VALU converts in
//     the GEMM inner loops).
// ---------------------------------------------------------------------------

#define E_   8192
#define N_   4096
#define ND_  64
#define ED_  16
#define H_   256
#define T_   4
#define CAP_ 32      // max neighbors kept per row (expected ~8; tail prob ~0)
#define KI_  96      // init GEMM K (80 padded to 3*32)
#define KA_  320     // agg GEMM K (64 + 256)

typedef __attribute__((ext_vector_type(16))) __bf16 v16bf;
typedef __attribute__((ext_vector_type(8)))  __bf16 v8bf;
typedef __attribute__((ext_vector_type(8)))  float  v8f;

__device__ __forceinline__ __bf16 f2bf(float f) {
  uint32_t u = __float_as_uint(f);
  u += 0x7fffu + ((u >> 16) & 1u);          // round-to-nearest-even
  union { uint16_t u16; __bf16 b; } cv;
  cv.u16 = (uint16_t)(u >> 16);
  return cv.b;
}

// Load one lane's share of a 16x32 bf16 WMMA operand from row-major storage.
// Lane (r = lane&15, half = lane>>4) holds row/col `r`, K-chunks
// [k0+8*half .. +7] and [k0+16+8*half .. +7]  (ISA 7.12.2 layout).
__device__ __forceinline__ v16bf ld_tile(const __bf16* rowptr, int k0, int half) {
  const v8bf lo = *(const v8bf*)(rowptr + k0 + half * 8);
  const v8bf hi = *(const v8bf*)(rowptr + k0 + half * 8 + 16);
  v16bf o;
#pragma unroll
  for (int j = 0; j < 8; ++j) { o[j] = lo[j]; o[j + 8] = hi[j]; }
  return o;
}

__device__ __forceinline__ v8f wmma_bf16(v16bf a, v16bf b, v8f c) {
  return __builtin_amdgcn_wmma_f32_16x16x32_bf16(
      /*neg_a=*/false, a, /*neg_b=*/false, b,
      /*c_mod=*/(short)0, c, /*reuse_a=*/false, /*reuse_b=*/false);
}

// --------------------------- weight prep (f32 -> bf16, transposed) ----------
__global__ void k_prep_init(const float* __restrict__ W, __bf16* __restrict__ wt) {
  int i = blockIdx.x * blockDim.x + threadIdx.x;          // H_*KI_
  if (i >= H_ * KI_) return;
  int n = i / KI_, k = i % KI_;
  float v = (k < ND_ + ED_) ? W[(size_t)k * H_ + n] : 0.f;
  wt[i] = f2bf(v);
}

__global__ void k_prep_pass(const float* __restrict__ W, __bf16* __restrict__ wt) {
  int i = blockIdx.x * blockDim.x + threadIdx.x;          // T_*H_*H_ exact
  int t = i >> 16, rem = i & 65535;
  int n = rem >> 8, k = rem & 255;
  wt[i] = f2bf(W[(size_t)t * 65536 + (size_t)k * H_ + n]);
}

__global__ void k_prep_agg(const float* __restrict__ W, __bf16* __restrict__ wt) {
  int i = blockIdx.x * blockDim.x + threadIdx.x;          // H_*KA_
  if (i >= H_ * KA_) return;
  int n = i / KA_, k = i % KA_;
  wt[i] = f2bf(W[(size_t)k * H_ + n]);
}

// --------------------------- A-panel prep (f32 -> bf16) ---------------------
// a0[E][96] = concat(edge_aligned_node_features, dir_edge_features, zero-pad)
__global__ void k_prep_a0(const float* __restrict__ eanf, const float* __restrict__ edf,
                          __bf16* __restrict__ a0) {
  int i = blockIdx.x * blockDim.x + threadIdx.x;          // E_*KI_ exact
  int row = i / KI_, k = i % KI_;
  float v;
  if (k < ND_)            v = eanf[(size_t)row * ND_ + k];
  else if (k < ND_ + ED_) v = edf[(size_t)row * ED_ + (k - ND_)];
  else                    v = 0.f;
  a0[i] = f2bf(v);
}

// aag[N][320] columns 0..63 = bf16(node_features); cols 64..319 filled by gather
__global__ void k_prep_nfcat(const float* __restrict__ nf, __bf16* __restrict__ aag) {
  int i = blockIdx.x * blockDim.x + threadIdx.x;          // N_*ND_ exact
  int row = i / ND_, k = i % ND_;
  aag[(size_t)row * KA_ + k] = f2bf(nf[i]);
}

// --------------------------- CSR build (deterministic, column-ordered) ------
// One 256-thread workgroup per row; thread owns a contiguous span of columns.
// Prefix-scan assigns slots in strict column order -> bitwise-stable output.
__global__ void k_csr(const float* __restrict__ adj, int cols,
                      int* __restrict__ idx, int* __restrict__ cnt) {
  const int row = blockIdx.x, tid = threadIdx.x;
  const int per4 = (cols >> 2) / 256;                     // float4 per thread
  const float4* r4 = (const float4*)(adj + (size_t)row * cols) + (size_t)tid * per4;

  unsigned my = 0;
  for (int i = 0; i < per4; ++i) {
    __builtin_prefetch((const void*)(r4 + i + per4), 0, 0);
    float4 v = r4[i];
    my += (v.x != 0.f) + (v.y != 0.f) + (v.z != 0.f) + (v.w != 0.f);
  }

  __shared__ unsigned sc[256];
  sc[tid] = my;
  __syncthreads();
  for (int off = 1; off < 256; off <<= 1) {               // Hillis-Steele scan
    unsigned v = (tid >= off) ? sc[tid - off] : 0u;
    __syncthreads();
    sc[tid] += v;
    __syncthreads();
  }
  unsigned wp = sc[tid] - my;                             // exclusive prefix
  unsigned total = sc[255];

  for (int i = 0; i < per4; ++i) {                        // row now L1/L2 hot
    float4 v = r4[i];
    int cb = (tid * per4 + i) * 4;
    if (v.x != 0.f) { if (wp < CAP_) idx[(size_t)row * CAP_ + wp] = cb + 0; ++wp; }
    if (v.y != 0.f) { if (wp < CAP_) idx[(size_t)row * CAP_ + wp] = cb + 1; ++wp; }
    if (v.z != 0.f) { if (wp < CAP_) idx[(size_t)row * CAP_ + wp] = cb + 2; ++wp; }
    if (v.w != 0.f) { if (wp < CAP_) idx[(size_t)row * CAP_ + wp] = cb + 3; ++wp; }
  }
  if (tid == 0) cnt[row] = (int)(total < CAP_ ? total : CAP_);
}

// --------------------------- gather + fused LayerNorm -> bf16 ---------------
// Writes LN'd row into out[row*ld_out + col_off + tid] (lets the node-side
// aggregation land directly inside the concat A-panel of the agg GEMM).
__global__ void k_gather_ln(const float* __restrict__ hin,
                            const int* __restrict__ idx, const int* __restrict__ cnt,
                            const float* __restrict__ g, const float* __restrict__ bia,
                            __bf16* __restrict__ out, int ld_out, int col_off) {
  const int row = blockIdx.x, tid = threadIdx.x;
  __shared__ int sidx[CAP_];
  __shared__ int scnt;
  if (tid == 0) scnt = cnt[row];
  if (tid < CAP_) sidx[tid] = idx[(size_t)row * CAP_ + tid];
  __syncthreads();

  float acc = 0.f;
  const int n = scnt;
  for (int i = 0; i < n; ++i)
    acc += hin[(size_t)sidx[i] * H_ + tid];

  __shared__ float s1[256], s2[256];
  s1[tid] = acc; s2[tid] = acc * acc;
  __syncthreads();
  for (int off = 128; off > 0; off >>= 1) {
    if (tid < off) { s1[tid] += s1[tid + off]; s2[tid] += s2[tid + off]; }
    __syncthreads();
  }
  float mean = s1[0] * (1.f / H_);
  float var  = s2[0] * (1.f / H_) - mean * mean;
  float y = (acc - mean) * rsqrtf(var + 1e-6f) * g[tid] + bia[tid];
  out[(size_t)row * ld_out + col_off + tid] = f2bf(y);
}

// --------------------------- WMMA GEMMs -------------------------------------
// One wave per 16x16 output tile; EXEC all-ones at every WMMA; all operands
// pre-converted bf16 -> inner loop is pure global_load_b128 + v_wmma.

__global__ void __launch_bounds__(256)
k_gemm_init(const __bf16* __restrict__ a0, const __bf16* __restrict__ wt,
            const float* __restrict__ bias,
            float* __restrict__ h0, float* __restrict__ h) {
  const int lane = threadIdx.x & 31, wid = threadIdx.x >> 5;
  const int tile = blockIdx.x * 8 + wid;
  const int m0 = (tile >> 4) * 16, n0 = (tile & 15) * 16;
  const int r = lane & 15, half = lane >> 4;
  const __bf16* arow = a0 + (size_t)(m0 + r) * KI_;
  const __bf16* brow = wt + (size_t)(n0 + r) * KI_;
  v8f c = {};
#pragma unroll
  for (int ks = 0; ks < 3; ++ks) {
    v16bf a = ld_tile(arow, ks * 32, half);
    v16bf b = ld_tile(brow, ks * 32, half);
    c = wmma_bf16(a, b, c);
  }
#pragma unroll
  for (int i = 0; i < 8; ++i) {
    int m = m0 + half * 8 + i, nn = n0 + r;
    float v = c[i] + bias[nn];
    v = v > 0.f ? v : 0.f;
    h0[(size_t)m * H_ + nn] = v;
    h [(size_t)m * H_ + nn] = v;
  }
}

__global__ void __launch_bounds__(256)
k_gemm_pass(const __bf16* __restrict__ mbf, const __bf16* __restrict__ wt,
            const float* __restrict__ h0, const float* __restrict__ bias,
            float* __restrict__ h) {
  const int lane = threadIdx.x & 31, wid = threadIdx.x >> 5;
  const int tile = blockIdx.x * 8 + wid;
  const int m0 = (tile >> 4) * 16, n0 = (tile & 15) * 16;
  const int r = lane & 15, half = lane >> 4;
  const __bf16* arow = mbf + (size_t)(m0 + r) * H_;
  const __bf16* brow = wt  + (size_t)(n0 + r) * H_;
  v8f c = {};
#pragma unroll
  for (int ks = 0; ks < 8; ++ks) {
    v16bf a = ld_tile(arow, ks * 32, half);
    v16bf b = ld_tile(brow, ks * 32, half);
    c = wmma_bf16(a, b, c);
  }
#pragma unroll
  for (int i = 0; i < 8; ++i) {
    int m = m0 + half * 8 + i, nn = n0 + r;
    float v = h0[(size_t)m * H_ + nn] + c[i] + bias[nn];
    v = v > 0.f ? v : 0.f;
    h[(size_t)m * H_ + nn] = v;
  }
}

__global__ void __launch_bounds__(256)
k_gemm_agg(const __bf16* __restrict__ aag, const __bf16* __restrict__ wt,
           const float* __restrict__ bias, float* __restrict__ hv) {
  const int lane = threadIdx.x & 31, wid = threadIdx.x >> 5;
  const int tile = blockIdx.x * 8 + wid;
  const int m0 = (tile >> 4) * 16, n0 = (tile & 15) * 16;
  const int r = lane & 15, half = lane >> 4;
  const __bf16* arow = aag + (size_t)(m0 + r) * KA_;
  const __bf16* brow = wt  + (size_t)(n0 + r) * KA_;
  v8f c = {};
#pragma unroll
  for (int ks = 0; ks < 10; ++ks) {
    v16bf a = ld_tile(arow, ks * 32, half);
    v16bf b = ld_tile(brow, ks * 32, half);
    c = wmma_bf16(a, b, c);
  }
#pragma unroll
  for (int i = 0; i < 8; ++i) {
    int m = m0 + half * 8 + i, nn = n0 + r;
    float v = c[i] + bias[nn];
    hv[(size_t)m * H_ + nn] = v > 0.f ? v : 0.f;
  }
}

// --------------------------- final LN + deterministic column sum ------------
__global__ void k_ln_rows(float* __restrict__ hv,
                          const float* __restrict__ g, const float* __restrict__ bia) {
  const int row = blockIdx.x, tid = threadIdx.x;
  float x = hv[(size_t)row * H_ + tid];
  __shared__ float s1[256], s2[256];
  s1[tid] = x; s2[tid] = x * x;
  __syncthreads();
  for (int off = 128; off > 0; off >>= 1) {
    if (tid < off) { s1[tid] += s1[tid + off]; s2[tid] += s2[tid + off]; }
    __syncthreads();
  }
  float mean = s1[0] * (1.f / H_);
  float var  = s2[0] * (1.f / H_) - mean * mean;
  hv[(size_t)row * H_ + tid] = (x - mean) * rsqrtf(var + 1e-6f) * g[tid] + bia[tid];
}

__global__ void k_colsum_part(const float* __restrict__ hv, float* __restrict__ part) {
  const int b = blockIdx.x, tid = threadIdx.x;      // 32 blocks x 128 rows
  float s = 0.f;
  for (int i = 0; i < N_ / 32; ++i)
    s += hv[(size_t)(b * (N_ / 32) + i) * H_ + tid];
  part[b * H_ + tid] = s;
}

__global__ void k_colsum_final(const float* __restrict__ part, float* __restrict__ out) {
  const int tid = threadIdx.x;
  float s = 0.f;
  for (int b = 0; b < 32; ++b) s += part[b * H_ + tid];
  out[tid] = s;
}

// --------------------------- launcher ---------------------------------------
extern "C" void kernel_launch(void* const* d_in, const int* in_sizes, int n_in,
                              void* d_out, int out_size, void* d_ws, size_t ws_size,
                              hipStream_t stream) {
  (void)in_sizes; (void)n_in; (void)out_size; (void)ws_size;
  const float* eanf   = (const float*)d_in[0];
  const float* edf    = (const float*)d_in[1];
  const float* nf     = (const float*)d_in[2];
  const float* adj_ee = (const float*)d_in[3];
  const float* adj_ne = (const float*)d_in[4];
  const float* W_init = (const float*)d_in[5];
  const float* b_init = (const float*)d_in[6];
  const float* W_pass = (const float*)d_in[7];
  const float* b_pass = (const float*)d_in[8];
  const float* ln_pg  = (const float*)d_in[9];
  const float* ln_pb  = (const float*)d_in[10];
  const float* W_agg  = (const float*)d_in[11];
  const float* b_agg  = (const float*)d_in[12];
  const float* ln_ag  = (const float*)d_in[13];
  const float* ln_ab  = (const float*)d_in[14];

  char* ws = (char*)d_ws;
  size_t off = 0;
  auto alloc = [&](size_t bytes) -> void* {
    void* p = ws + off;
    off = (off + bytes + 255) & ~(size_t)255;
    return p;
  };
  __bf16* wt_init = (__bf16*)alloc((size_t)H_ * KI_ * 2);
  __bf16* wt_pass = (__bf16*)alloc((size_t)T_ * H_ * H_ * 2);
  __bf16* wt_agg  = (__bf16*)alloc((size_t)H_ * KA_ * 2);
  __bf16* a0      = (__bf16*)alloc((size_t)E_ * KI_ * 2);
  __bf16* aag     = (__bf16*)alloc((size_t)N_ * KA_ * 2);
  int*    ee_idx  = (int*)alloc((size_t)E_ * CAP_ * 4);
  int*    ee_cnt  = (int*)alloc((size_t)E_ * 4);
  int*    ne_idx  = (int*)alloc((size_t)N_ * CAP_ * 4);
  int*    ne_cnt  = (int*)alloc((size_t)N_ * 4);
  float*  h0      = (float*)alloc((size_t)E_ * H_ * 4);
  float*  h       = (float*)alloc((size_t)E_ * H_ * 4);
  __bf16* mbf     = (__bf16*)alloc((size_t)E_ * H_ * 2);
  float*  hv      = (float*)alloc((size_t)N_ * H_ * 4);
  float*  part    = (float*)alloc((size_t)32 * H_ * 4);

  // weight + A-panel conversion (tiny, once per launch)
  k_prep_init <<<(H_ * KI_ + 255) / 256, 256, 0, stream>>>(W_init, wt_init);
  k_prep_pass <<<(T_ * H_ * H_) / 256, 256, 0, stream>>>(W_pass, wt_pass);
  k_prep_agg  <<<(H_ * KA_ + 255) / 256, 256, 0, stream>>>(W_agg, wt_agg);
  k_prep_a0   <<<(E_ * KI_) / 256, 256, 0, stream>>>(eanf, edf, a0);
  k_prep_nfcat<<<(N_ * ND_) / 256, 256, 0, stream>>>(nf, aag);

  // one-shot sparse extraction of both adjacencies (streams 384 MB total)
  k_csr<<<E_, 256, 0, stream>>>(adj_ee, E_, ee_idx, ee_cnt);
  k_csr<<<N_, 256, 0, stream>>>(adj_ne, E_, ne_idx, ne_cnt);

  // h0 = relu(concat @ W_init + b);  h = h0
  k_gemm_init<<<(E_ / 16) * (H_ / 16) / 8, 256, 0, stream>>>(a0, wt_init, b_init, h0, h);

  // T message-passing steps: gather+LN (bf16) then WMMA GEMM with residual
  for (int t = 0; t < T_; ++t) {
    k_gather_ln<<<E_, 256, 0, stream>>>(h, ee_idx, ee_cnt, ln_pg, ln_pb, mbf, H_, 0);
    k_gemm_pass<<<(E_ / 16) * (H_ / 16) / 8, 256, 0, stream>>>(
        mbf, wt_pass + (size_t)t * H_ * H_, h0, b_pass + t * H_, h);
  }

  // node aggregation lands directly in the concat A-panel, then agg GEMM
  k_gather_ln<<<N_, 256, 0, stream>>>(h, ne_idx, ne_cnt, ln_ag, ln_ab, aag, KA_, ND_);
  k_gemm_agg<<<(N_ / 16) * (H_ / 16) / 8, 256, 0, stream>>>(aag, wt_agg, b_agg, hv);
  k_ln_rows<<<N_, 256, 0, stream>>>(hv, ln_ag, ln_ab);
  k_colsum_part<<<32, 256, 0, stream>>>(hv, part);
  k_colsum_final<<<1, 256, 0, stream>>>(part, (float*)d_out);
}
